// ContextualLoss_binary_66589172957474
// MI455X (gfx1250) — matrix operand order repaired and checked by reading, not compiled, not run on hardware
//
#include <hip/hip_runtime.h>
#include <hip/hip_bf16.h>

// ContextualLoss, fused for MI455X (gfx1250, wave32, WMMA f16->f32).
// B=4, C=64, N=4096. d (256MB) is never materialized: per 16-row tile we do
// two WMMA sweeps over all 4096 columns (min pass, then exp-sum pass).

#define B_   4
#define C_   64
#define N_   4096
#define EPSN 1e-10f
#define EPSM 0.001f

typedef __attribute__((ext_vector_type(16))) _Float16 v16h;
typedef __attribute__((ext_vector_type(8)))  _Float16 v8h;
typedef __attribute__((ext_vector_type(8)))  float    v8f;

// ---- order-preserving float<->uint encode for atomicMin on float bits ----
__device__ __forceinline__ unsigned fenc(float f) {
    unsigned u = __float_as_uint(f);
    return u ^ ((unsigned)((int)u >> 31) | 0x80000000u);
}
__device__ __forceinline__ float fdec(unsigned u) {
    unsigned m = (u & 0x80000000u) ? 0x80000000u : 0xFFFFFFFFu;
    return __uint_as_float(u ^ m);
}

// ---- kernel 1: spatial mean of Y per (b,c); also zero the CX accumulators --
__global__ __launch_bounds__(256) void cxl_mean_init(const float* __restrict__ Y,
                                                     float* __restrict__ mean,
                                                     float* __restrict__ cx) {
    __shared__ float red[256];
    int bc = blockIdx.x;                       // 0 .. B*C-1
    const float* row = Y + (size_t)bc * N_;
    float s = 0.0f;
    for (int i = threadIdx.x; i < N_; i += 256) s += row[i];
    red[threadIdx.x] = s;
    __syncthreads();
    for (int off = 128; off > 0; off >>= 1) {
        if (threadIdx.x < off) red[threadIdx.x] += red[threadIdx.x + off];
        __syncthreads();
    }
    if (threadIdx.x == 0) mean[bc] = red[0] * (1.0f / N_);
    if (threadIdx.x == 1 && bc < B_) cx[bc] = 0.0f;
}

// ---- kernel 2: center by y_mean, L2-normalize over C, transpose to [B,N,C] f16
__global__ __launch_bounds__(256) void cxl_normalize(const float* __restrict__ X,
                                                     const float* __restrict__ Y,
                                                     const float* __restrict__ mean,
                                                     _Float16* __restrict__ Xh,
                                                     _Float16* __restrict__ Yh) {
    int g = blockIdx.x * 256 + threadIdx.x;    // 0 .. B*N-1
    int b = g >> 12;                           // /4096
    int n = g & (N_ - 1);
    const float* xb = X + (size_t)b * C_ * N_ + n;
    const float* yb = Y + (size_t)b * C_ * N_ + n;
    const float* mb = mean + b * C_;
    float sx = 0.0f, sy = 0.0f;
    #pragma unroll 8
    for (int c = 0; c < C_; ++c) {
        float m  = mb[c];
        float xv = xb[(size_t)c * N_] - m;
        float yv = yb[(size_t)c * N_] - m;
        sx = fmaf(xv, xv, sx);
        sy = fmaf(yv, yv, sy);
    }
    float rx = 1.0f / (sqrtf(sx) + EPSN);
    float ry = 1.0f / (sqrtf(sy) + EPSN);
    _Float16* xo = Xh + (size_t)g * C_;
    _Float16* yo = Yh + (size_t)g * C_;
    #pragma unroll 8
    for (int c = 0; c < C_; ++c) {
        float m = mb[c];
        xo[c] = (_Float16)((xb[(size_t)c * N_] - m) * rx);
        yo[c] = (_Float16)((yb[(size_t)c * N_] - m) * ry);
    }
}

// ---- 16x16 tile of <x_i, y_j> via two v_wmma_f32_16x16x32_f16 (K = 64) ----
__device__ __forceinline__ v8f cxl_tile_dot(const _Float16* __restrict__ Yb,
                                            v16h a_lo, v16h a_hi,
                                            int col, int kh) {
    const _Float16* yr = Yb + (size_t)col * C_;
    v16h b_lo = *(const v16h*)(yr + kh);        // K = kh .. kh+15   (32B load)
    v16h b_hi = *(const v16h*)(yr + 32 + kh);   // K = 32+kh .. 47+kh
    v8f acc = {};
    acc = __builtin_amdgcn_wmma_f32_16x16x32_f16(false, a_lo, false, b_lo,
                                                 (short)0, acc, false, false);
    acc = __builtin_amdgcn_wmma_f32_16x16x32_f16(false, a_hi, false, b_hi,
                                                 (short)0, acc, false, false);
    return acc;
}

// ---- kernel 3: fused d = 1 - X^T Y, row-min, exp-softmax row-sum, CX accum -
// grid (N/16, B), 256 threads = 8 waves; wave w covers columns [512w, 512w+512)
__global__ __launch_bounds__(256) void cxl_cx(const _Float16* __restrict__ Xh,
                                              const _Float16* __restrict__ Yh,
                                              float* __restrict__ cx) {
    __shared__ unsigned row_min[16];
    __shared__ float    row_sum[16];

    const int tid  = threadIdx.x;
    const int wave = tid >> 5;
    const int lane = tid & 31;
    const int b    = blockIdx.y;
    const int r0   = blockIdx.x * 16;
    const int mrow = lane & 15;
    const int hi   = lane >> 4;

    // A tile (16x64 f16), contiguous K layout: lanes 0-15 hold K {0..15},
    // lanes 16-31 hold K {16..31} (plus +32 for the second WMMA).
    const _Float16* xr = Xh + (size_t)(b * N_ + r0 + mrow) * C_;
    const int ka = hi * 16;
    v8h p0 = *(const v8h*)(xr + ka);
    v8h p1 = *(const v8h*)(xr + 8 + ka);
    v8h p2 = *(const v8h*)(xr + 32 + ka);
    v8h p3 = *(const v8h*)(xr + 40 + ka);
    v16h a_lo = __builtin_shufflevector(p0, p1, 0,1,2,3,4,5,6,7,8,9,10,11,12,13,14,15);
    v16h a_hi = __builtin_shufflevector(p2, p3, 0,1,2,3,4,5,6,7,8,9,10,11,12,13,14,15);

    const int kh = hi * 16;                       // B layout K-half selector
    const _Float16* Yb = Yh + (size_t)b * N_ * C_;

    if (tid < 16) { row_min[tid] = 0xFFFFFFFFu; row_sum[tid] = 0.0f; }
    __syncthreads();

    // ---- pass 1: row minima of d over all 4096 columns ----
    float vmin[8];
    #pragma unroll
    for (int k = 0; k < 8; ++k) vmin[k] = 3.0e38f;
    for (int t = 0; t < 32; ++t) {
        int colb = (wave * 32 + t) * 16;
        v8f acc = cxl_tile_dot(Yb, a_lo, a_hi, colb + mrow, kh);
        #pragma unroll
        for (int k = 0; k < 8; ++k) {
            float d = 1.0f - acc[k];
            vmin[k] = fminf(vmin[k], d);
        }
    }
    #pragma unroll
    for (int k = 0; k < 8; ++k)
        atomicMin(&row_min[hi * 8 + k], fenc(vmin[k]));
    __syncthreads();

    // ---- pass 2: recompute d, accumulate sum_j exp(0.2 - d/(5(m+eps))) ----
    float inv[8], vsum[8];
    #pragma unroll
    for (int k = 0; k < 8; ++k) {
        float m = fdec(row_min[hi * 8 + k]);
        inv[k]  = 1.0f / (5.0f * (m + EPSM));
        vsum[k] = 0.0f;
    }
    for (int t = 0; t < 32; ++t) {
        int colb = (wave * 32 + t) * 16;
        v8f acc = cxl_tile_dot(Yb, a_lo, a_hi, colb + mrow, kh);
        #pragma unroll
        for (int k = 0; k < 8; ++k) {
            float d = 1.0f - acc[k];
            vsum[k] += __expf(0.2f - d * inv[k]);
        }
    }
    #pragma unroll
    for (int k = 0; k < 8; ++k)
        atomicAdd(&row_sum[hi * 8 + k], vsum[k]);
    __syncthreads();

    // max_j A = exp at the row minimum / row sum; accumulate mean over rows
    if (tid < 16) {
        float m    = fdec(row_min[tid]);
        float iv   = 1.0f / (5.0f * (m + EPSM));
        float amax = __expf(0.2f - m * iv) / row_sum[tid];
        atomicAdd(&cx[b], amax);
    }
}

// ---- kernel 4: out[b] = -log(mean_i A_max) ----
__global__ void cxl_final(const float* __restrict__ cx, float* __restrict__ out) {
    int bIdx = threadIdx.x;
    if (bIdx < B_) out[bIdx] = -logf(cx[bIdx] * (1.0f / N_));
}

extern "C" void kernel_launch(void* const* d_in, const int* in_sizes, int n_in,
                              void* d_out, int out_size, void* d_ws, size_t ws_size,
                              hipStream_t stream) {
    (void)in_sizes; (void)n_in; (void)out_size; (void)ws_size;
    const float* X = (const float*)d_in[0];
    const float* Y = (const float*)d_in[1];
    float* out = (float*)d_out;

    // workspace layout: mean[256] f32 | cx[4] f32 | pad | Xh[B*N*C] f16 | Yh[...]
    float*    mean = (float*)d_ws;
    float*    cx   = mean + 256;
    _Float16* Xh   = (_Float16*)((char*)d_ws + 2048);
    _Float16* Yh   = Xh + (size_t)B_ * N_ * C_;

    cxl_mean_init<<<B_ * C_, 256, 0, stream>>>(Y, mean, cx);
    cxl_normalize<<<(B_ * N_) / 256, 256, 0, stream>>>(X, Y, mean, Xh, Yh);
    dim3 grid(N_ / 16, B_);
    cxl_cx<<<grid, 256, 0, stream>>>(Xh, Yh, cx);
    cxl_final<<<1, 32, 0, stream>>>(cx, out);
}
